// Seq2Seq_24232205484356
// MI455X (gfx1250) — compile-verified, matrix operand order checked by reading, not run on hardware
//
#include <hip/hip_runtime.h>

// Problem constants (reference: B=64, T=64, E=64, H=128, VIN=256, VOUT=32000)
#define B_    64
#define T_    64
#define E_    64
#define H_    128
#define G4H   512        // 4*H
#define VOUT_ 32000

typedef _Float16 v16h __attribute__((ext_vector_type(16)));
typedef _Float16 v8h  __attribute__((ext_vector_type(8)));
typedef _Float16 v4h  __attribute__((ext_vector_type(4)));
typedef float    v8f  __attribute__((ext_vector_type(8)));
typedef float    v4f  __attribute__((ext_vector_type(4)));

// ---- fast branch-free activations on the TRANS unit (critical for the
// serial LSTM recurrence: no EXEC-divergent libm expansions).
__device__ __forceinline__ float fast_exp2(float x) { return __builtin_amdgcn_exp2f(x); }
__device__ __forceinline__ float fast_rcp(float x)  { return __builtin_amdgcn_rcpf(x); }

__device__ __forceinline__ float sigmf(float x) {
  // 1/(1+e^-x); exp2->inf => rcp->0, exp2->0 => 1 : edge cases are graceful.
  return fast_rcp(1.0f + fast_exp2(-1.44269504f * x));
}

__device__ __forceinline__ float tanh_fast(float x) {
#if __has_builtin(__builtin_amdgcn_tanhf)
  return __builtin_amdgcn_tanhf(x);       // gfx125x V_TANH_F32
#else
  float xc = fminf(fmaxf(x, -9.0f), 9.0f);          // avoid inf/inf
  float e2 = fast_exp2(2.88539008f * xc);           // e^(2x)
  return (e2 - 1.0f) * fast_rcp(e2 + 1.0f);
#endif
}

// Load a 16-half WMMA A/B fragment: two contiguous 16B runs at p and p+16
// (ISA 16-bit operand layout: lane covers K=[hi*8,+8) and [16+hi*8,+8)).
__device__ __forceinline__ v16h ld_frag(const _Float16* p) {
  v8h lo = *(const v8h*)p;
  v8h hi = *(const v8h*)(p + 16);
  return __builtin_shufflevector(lo, hi, 0, 1, 2, 3, 4, 5, 6, 7,
                                 8, 9, 10, 11, 12, 13, 14, 15);
}

// ---------------------------------------------------------------- utilities
__global__ void k_f32_to_f16(const float* __restrict__ s, _Float16* __restrict__ d, int n) {
  int i = blockIdx.x * blockDim.x + threadIdx.x;
  if (i < n) d[i] = (_Float16)s[i];
}

__global__ void k_zero_t0(float* __restrict__ out) {
  int i = blockIdx.x * blockDim.x + threadIdx.x;   // exactly 64*32000 threads
  int b = i / VOUT_;
  int v = i - b * VOUT_;
  __builtin_nontemporal_store(0.0f, out + (size_t)b * (T_ * VOUT_) + v);  // out[b][0][v]
}

// ------------------------------------------------- xg = emb[tok] @ Wih^T + bih + bhh
// grid = Tsteps*B blocks (bid = t*64 + b), 128 threads each.
__global__ void k_embed_xg(const int* __restrict__ tok, int tok_stride,
                           const float* __restrict__ emb,
                           const float* __restrict__ Wih,
                           const float* __restrict__ bih,
                           const float* __restrict__ bhh,
                           float* __restrict__ xg) {
  const int bid = blockIdx.x;
  const int t = bid >> 6, b = bid & 63;
  __shared__ float xs[E_];
  const int tid = threadIdx.x;
  const int token = tok[b * tok_stride + t];
  if (tid < E_)
    xs[tid] = (token == 0) ? 0.0f : emb[(size_t)token * E_ + tid];  // padding_idx=0
  __syncthreads();
  float* orow = xg + (size_t)bid * G4H;
  #pragma unroll
  for (int j = 0; j < 4; ++j) {
    const int c = tid * 4 + j;
    const float* w = Wih + (size_t)c * E_;
    float s = bih[c] + bhh[c];
    #pragma unroll
    for (int k = 0; k < E_; ++k) s = fmaf(xs[k], w[k], s);
    orow[c] = s;
  }
}

// ------------------------------------------------------------- LSTM recurrence
// One workgroup per 16 batch rows (grid = 4, block = 512 = 16 waves).
// Whh (B operand) is loop-invariant: preloaded once into 64 VGPRs/wave.
// Per step: gates[16,512] = h16[16,128] @ Whh16^T via v_wmma_f32_16x16x32_f16,
// then fused branch-free elementwise cell update. h in LDS (f16), c in regs.
__global__ void __launch_bounds__(512, 1)
k_lstm(const float* __restrict__ xg,
       const _Float16* __restrict__ Whh16,
       const float* __restrict__ h0, const float* __restrict__ c0,
       float* __restrict__ hN, float* __restrict__ cN,
       _Float16* __restrict__ hs16, int Tsteps) {
  __shared__ __align__(16) _Float16 hsh[16][H_];   // current h, f16 (A operand)
  __shared__ float gl[16][G4H];                    // gates staging
  const int tid  = threadIdx.x;
  const int lane = tid & 31, w = tid >> 5;         // 16 waves
  const int m = lane & 15, hi = lane >> 4;
  const int bbase = blockIdx.x * 16;

  // ---- preload loop-invariant B fragments: wave w owns N-tiles w*2, w*2+1
  v16h bfr[2][4];
  #pragma unroll
  for (int i = 0; i < 2; ++i) {
    const int n0 = (w * 2 + i) * 16;
    const _Float16* brow = Whh16 + (size_t)(n0 + m) * H_;
    #pragma unroll
    for (int kk = 0; kk < 4; ++kk)
      bfr[i][kk] = ld_frag(brow + kk * 32 + hi * 8);
  }

  // each thread owns 4 consecutive cell-state elements (single row segment)
  const int idx0 = tid * 4;
  const int row0 = idx0 >> 7;          // 128 cols per row, 4 | 128 => same row
  const int col0 = idx0 & 127;
  float creg[4];
  #pragma unroll
  for (int e = 0; e < 4; ++e) {
    creg[e] = c0 ? c0[(size_t)(bbase + row0) * H_ + col0 + e] : 0.0f;
    hsh[row0][col0 + e] =
        h0 ? (_Float16)h0[(size_t)(bbase + row0) * H_ + col0 + e] : (_Float16)0.0f;
  }
  __syncthreads();

  for (int t = 0; t < Tsteps; ++t) {
    // ---- prefetch this step's xg (independent of h; overlaps the GEMM)
    const float* xrow = xg + ((size_t)t * B_ + bbase + row0) * G4H;
    v4f xgv[4];
    #pragma unroll
    for (int g = 0; g < 4; ++g)
      xgv[g] = *(const v4f*)(xrow + g * 128 + col0);

    // ---- GEMM: gates = h @ Whh^T (K = 128, four K=32 WMMA steps)
    v16h afrag[4];
    #pragma unroll
    for (int kk = 0; kk < 4; ++kk)
      afrag[kk] = ld_frag(&hsh[m][kk * 32 + hi * 8]);

    #pragma unroll
    for (int i = 0; i < 2; ++i) {                  // 2 N-tiles per wave
      const int n0 = (w * 2 + i) * 16;
      v8f acc = {};
      #pragma unroll
      for (int kk = 0; kk < 4; ++kk)
        acc = __builtin_amdgcn_wmma_f32_16x16x32_f16(
            false, afrag[kk], false, bfr[i][kk], (short)0, acc, false, false);
      #pragma unroll
      for (int r = 0; r < 8; ++r)                  // D layout: row = r + hi*8, col = m
        gl[hi * 8 + r][n0 + m] = acc[r];
    }
    __syncthreads();

    // ---- elementwise cell update (torch gate order i,f,g,o), branch-free
    v4h hpack;
    #pragma unroll
    for (int e = 0; e < 4; ++e) {
      const int c = col0 + e;
      float gi = gl[row0][c]       + xgv[0][e];
      float gf = gl[row0][128 + c] + xgv[1][e];
      float gg = gl[row0][256 + c] + xgv[2][e];
      float go = gl[row0][384 + c] + xgv[3][e];
      float cc = sigmf(gf) * creg[e] + sigmf(gi) * tanh_fast(gg);
      creg[e] = cc;
      float hh = sigmf(go) * tanh_fast(cc);
      hpack[e] = (_Float16)hh;
    }
    *(v4h*)&hsh[row0][col0] = hpack;               // 8B LDS store
    if (hs16)                                      // 8B global store
      *(v4h*)(hs16 + ((size_t)t * B_ + bbase + row0) * H_ + col0) = hpack;
    __syncthreads();
  }

  // ---- final state (encoder only), off the per-step critical path
  if (hN) {
    #pragma unroll
    for (int e = 0; e < 4; ++e) {
      hN[(size_t)(bbase + row0) * H_ + col0 + e] = (float)hsh[row0][col0 + e];
      cN[(size_t)(bbase + row0) * H_ + col0 + e] = creg[e];
    }
  }
}

// ----------------------------------------------------- FC: logits = hs @ fc_W^T + b
// M = 63*64 = 4032, N = 32000, K = 128. Each wave computes a 64x64 tile
// (16 f32 accumulators, 64 WMMAs). Output via nontemporal stores (512 MB,
// HBM-store-bound: keep L2 for the 8 MB f16 weight matrix).
__global__ void __launch_bounds__(128, 1)
k_fc(const _Float16* __restrict__ hs16,
     const _Float16* __restrict__ W16,
     const float* __restrict__ bias,
     float* __restrict__ out) {
  const int tid  = threadIdx.x;
  const int lane = tid & 31, w = tid >> 5;
  const int m = lane & 15, hi = lane >> 4;
  const int strip = blockIdx.x * 4 + w;            // blockDim = 128 (4 waves)
  const int sm = strip / 500;                      // 0..62  (M strips of 64)
  const int sn = strip - sm * 500;                 // 0..499 (N strips of 64)
  const int mbase = sm * 64;
  const int nbase = sn * 64;

  v8f zero = {};
  v8f acc[4][4];
  #pragma unroll
  for (int mt = 0; mt < 4; ++mt)
    #pragma unroll
    for (int nt = 0; nt < 4; ++nt) acc[mt][nt] = zero;

  #pragma unroll
  for (int kk = 0; kk < 4; ++kk) {                 // K = 4 x 32
    v16h a[4], bm[4];
    #pragma unroll
    for (int mt = 0; mt < 4; ++mt)
      a[mt] = ld_frag(hs16 + (size_t)(mbase + mt * 16 + m) * H_ + kk * 32 + hi * 8);
    #pragma unroll
    for (int nt = 0; nt < 4; ++nt)
      bm[nt] = ld_frag(W16 + (size_t)(nbase + nt * 16 + m) * H_ + kk * 32 + hi * 8);
    #pragma unroll
    for (int mt = 0; mt < 4; ++mt)
      #pragma unroll
      for (int nt = 0; nt < 4; ++nt)
        acc[mt][nt] = __builtin_amdgcn_wmma_f32_16x16x32_f16(
            false, a[mt], false, bm[nt], (short)0, acc[mt][nt], false, false);
  }

  // epilogue: row sm*64 + j maps to (ti = sm, b = j); store to out[b][ti+1][n]
  const int ti = sm;
  #pragma unroll
  for (int nt = 0; nt < 4; ++nt) {
    const int n = nbase + nt * 16 + m;
    const float bv = bias[n];
    #pragma unroll
    for (int mt = 0; mt < 4; ++mt) {
      #pragma unroll
      for (int r = 0; r < 8; ++r) {
        const int b = mt * 16 + hi * 8 + r;
        float v = acc[mt][nt][r] + bv;
        __builtin_nontemporal_store(
            v, out + (size_t)b * (T_ * VOUT_) + (size_t)(ti + 1) * VOUT_ + n);
      }
    }
  }
}

// ---------------------------------------------------------------------------
extern "C" void kernel_launch(void* const* d_in, const int* in_sizes, int n_in,
                              void* d_out, int out_size, void* d_ws, size_t ws_size,
                              hipStream_t stream) {
  (void)in_sizes; (void)n_in; (void)out_size; (void)ws_size;
  const int*   src     = (const int*)d_in[0];
  const int*   tgt     = (const int*)d_in[1];
  const float* enc_emb = (const float*)d_in[2];
  const float* dec_emb = (const float*)d_in[3];
  const float* enc_Wih = (const float*)d_in[4];
  const float* enc_Whh = (const float*)d_in[5];
  const float* enc_bih = (const float*)d_in[6];
  const float* enc_bhh = (const float*)d_in[7];
  const float* dec_Wih = (const float*)d_in[8];
  const float* dec_Whh = (const float*)d_in[9];
  const float* dec_bih = (const float*)d_in[10];
  const float* dec_bhh = (const float*)d_in[11];
  const float* fc_W    = (const float*)d_in[12];
  const float* fc_b    = (const float*)d_in[13];
  float* out = (float*)d_out;

  // workspace partition (~26 MB)
  char* ws = (char*)d_ws;
  size_t off = 0;
  auto alloc = [&](size_t bytes) {
    void* p = ws + off;
    off = (off + bytes + 255) & ~(size_t)255;
    return p;
  };
  float*    xg_enc   = (float*)alloc((size_t)T_ * B_ * G4H * 4);
  float*    xg_dec   = (float*)alloc((size_t)(T_ - 1) * B_ * G4H * 4);
  _Float16* encWhh16 = (_Float16*)alloc((size_t)G4H * H_ * 2);
  _Float16* decWhh16 = (_Float16*)alloc((size_t)G4H * H_ * 2);
  _Float16* fcW16    = (_Float16*)alloc((size_t)VOUT_ * H_ * 2);
  _Float16* hs16     = (_Float16*)alloc((size_t)(T_ - 1) * B_ * H_ * 2);
  float*    h0       = (float*)alloc((size_t)B_ * H_ * 4);
  float*    c0       = (float*)alloc((size_t)B_ * H_ * 4);

  // weight conversions (fc_W16 = 8 MB -> resident in 192 MB L2 for the GEMM)
  k_f32_to_f16<<<(G4H * H_ + 255) / 256, 256, 0, stream>>>(enc_Whh, encWhh16, G4H * H_);
  k_f32_to_f16<<<(G4H * H_ + 255) / 256, 256, 0, stream>>>(dec_Whh, decWhh16, G4H * H_);
  k_f32_to_f16<<<(VOUT_ * H_ + 255) / 256, 256, 0, stream>>>(fc_W, fcW16, VOUT_ * H_);

  // out[:, 0, :] = 0
  k_zero_t0<<<(B_ * VOUT_) / 256, 256, 0, stream>>>(out);

  // gate pre-compute (xg includes bih + bhh)
  k_embed_xg<<<T_ * B_, 128, 0, stream>>>(src, T_, enc_emb, enc_Wih, enc_bih, enc_bhh, xg_enc);
  k_embed_xg<<<(T_ - 1) * B_, 128, 0, stream>>>(tgt, T_, dec_emb, dec_Wih, dec_bih, dec_bhh, xg_dec);

  // recurrences: encoder -> (h0,c0); decoder (teacher forcing) -> hs16
  k_lstm<<<B_ / 16, 512, 0, stream>>>(xg_enc, encWhh16, nullptr, nullptr, h0, c0, nullptr, T_);
  k_lstm<<<B_ / 16, 512, 0, stream>>>(xg_dec, decWhh16, h0, c0, nullptr, nullptr, hs16, T_ - 1);

  // big GEMM: 63*500 = 31500 wave-strips, 4 waves/block
  k_fc<<<(63 * 500) / 4, 128, 0, stream>>>(hs16, fcW16, fc_b, out);
}
